// ReRankTransformer_38628935860482
// MI455X (gfx1250) — matrix operand order repaired and checked by reading, not compiled, not run on hardware
//
#include <hip/hip_runtime.h>

// ---------------------------------------------------------------------------
// Problem constants (from reference)
// ---------------------------------------------------------------------------
constexpr int kB     = 256;      // batch
constexpr int kC     = 128;      // channel
constexpr int kNN    = 500000;   // NUM_NODES
constexpr int kNID   = 50000;    // N_ID
constexpr int kTOPK  = 100;
constexpr int kD     = 256;      // 2*C
constexpr int kHEADS = 4;
constexpr int kHD    = 64;       // head dim
constexpr int kROWS  = kB * kTOPK;   // 25600 transformer rows
constexpr int kCAP   = 4096;     // top-k candidate capacity per row

// ---------------------------------------------------------------------------
// WMMA vector types (gfx1250, wave32)
// ---------------------------------------------------------------------------
typedef __attribute__((ext_vector_type(16))) __bf16 v16bf;
typedef __attribute__((ext_vector_type(2)))  __bf16 v2bf;
typedef __attribute__((ext_vector_type(8)))  float  v8f;

struct U8 { uint4 lo, hi; };   // 32 bytes == one v16bf fragment

// pack two floats into one dword of bf16 pair {lo=a, hi=b}; lowers to
// v_cvt_pk_bf16_f32 (RNE)
__device__ __forceinline__ unsigned pack_bf2(float a, float b) {
    v2bf v;
    v.x = (__bf16)a;
    v.y = (__bf16)b;
    return __builtin_bit_cast(unsigned, v);
}

__device__ __forceinline__ unsigned mono(float f) {
    unsigned u = __float_as_uint(f);
    return (u & 0x80000000u) ? ~u : (u | 0x80000000u);
}

// ---------------------------------------------------------------------------
// bf16-WMMA GEMM:  C[M,N] = A[M,K] * Bt[N,K]^T  (+ biases / epilogue)
//
// Block 256 threads (8 waves as 4x2). Block tile 128x64, K-chunk 32.
// Each wave owns a 32x32 tile: 2 A-frags x 2 B-frags -> 4 WMMAs w/ reuse.
// LDS holds bf16 pairs packed in dwords; row stride 20 dwords (80B) keeps
// 16B alignment for b128 access and spreads fragment lanes across banks.
// Requirements: M % 128 == 0, K % 32 == 0 (true for all call sites).
// N handled by clamping B-row addresses (padding columns are computed but
// never stored: epilogue guards gn < N).
// mode 0: C = acc + biases
// mode 1: C = Res + relu(acc + colBias)      (FFN)
// mode 2: C = Res + acc + colBias            (residual add)
// ---------------------------------------------------------------------------
__global__ __launch_bounds__(256) void k_gemm_bf16(
    const float* __restrict__ A, const float* __restrict__ Bt,
    const float* __restrict__ rowBias, const float* __restrict__ colBias,
    const float* __restrict__ Res, float* __restrict__ C,
    int M, int N, int K, int mode)
{
    __shared__ __align__(16) unsigned As[128][20];  // 128 rows x 16 dwords (+4 pad)
    __shared__ __align__(16) unsigned Bs[64][20];

    const int tid   = threadIdx.x;
    const int lane  = tid & 31;
    const int wid   = tid >> 5;
    const int waveM = wid >> 1;          // 0..3  -> 32-row slice
    const int waveN = wid & 1;           // 0..1  -> 32-col slice
    const int half  = lane >> 4;
    const int l15   = lane & 15;
    const int m0    = blockIdx.y * 128;
    const int n0    = blockIdx.x * 64;

    // staging coordinates: one 8-float K-chunk per (row, quarter)
    const int srow = tid >> 2;           // 0..63
    const int skc  = (tid & 3) * 8;      // K element offset 0,8,16,24

    // clamped B row: padding rows produce garbage only in discarded columns
    const int gn_clamped = min(n0 + srow, N - 1);

    v8f acc[2][2] = {{{}, {}}, {{}, {}}};

    for (int k0 = 0; k0 < K; k0 += 32) {
        // ---- stage A (128x32) : two 64-row halves, b128 loads + b128 LDS stores
        #pragma unroll
        for (int it = 0; it < 2; ++it) {
            const int r = srow + it * 64;
            const float* p = A + (size_t)(m0 + r) * K + k0 + skc;
            const float4 f0 = *(const float4*)p;
            const float4 f1 = *(const float4*)(p + 4);
            uint4 q;
            q.x = pack_bf2(f0.x, f0.y);
            q.y = pack_bf2(f0.z, f0.w);
            q.z = pack_bf2(f1.x, f1.y);
            q.w = pack_bf2(f1.z, f1.w);
            *(uint4*)&As[r][skc >> 1] = q;
        }
        // ---- stage B (64x32), unconditional via clamped row address
        {
            const float* p = Bt + (size_t)gn_clamped * K + k0 + skc;
            const float4 f0 = *(const float4*)p;
            const float4 f1 = *(const float4*)(p + 4);
            uint4 q;
            q.x = pack_bf2(f0.x, f0.y);
            q.y = pack_bf2(f0.z, f0.w);
            q.z = pack_bf2(f1.x, f1.y);
            q.w = pack_bf2(f1.z, f1.w);
            *(uint4*)&Bs[srow][skc >> 1] = q;
        }
        __syncthreads();

        // prefetch next weight chunk (emits global_prefetch_b8)
        if (k0 + 32 < K)
            __builtin_prefetch(Bt + (size_t)gn_clamped * K + k0 + 32 + skc, 0, 1);

        // ---- fragment assembly: 2x ds_load_b128 per fragment
        // A 16x32 bf16, ISA lane layout: lane(l15,half) -> row, K pairs
        //   pair idx = half*4 + {0..3} and half*4 + 8 + {0..3}
        v16bf af[2], bfr[2];
        #pragma unroll
        for (int mi = 0; mi < 2; ++mi) {
            const unsigned* Ar = As[waveM * 32 + mi * 16 + l15];
            U8 u;
            u.lo = *(const uint4*)(Ar + half * 4);
            u.hi = *(const uint4*)(Ar + half * 4 + 8);
            af[mi] = __builtin_bit_cast(v16bf, u);
        }
        // B 32x16 bf16: lane -> col, K pairs = half*4 + {0..3} and half*4 + 8 + {0..3}
        #pragma unroll
        for (int s = 0; s < 2; ++s) {
            const unsigned* Br = Bs[waveN * 32 + s * 16 + l15];
            U8 u;
            u.lo = *(const uint4*)(Br + half * 4);
            u.hi = *(const uint4*)(Br + half * 4 + 8);
            bfr[s] = __builtin_bit_cast(v16bf, u);
        }

        #pragma unroll
        for (int mi = 0; mi < 2; ++mi)
            #pragma unroll
            for (int s = 0; s < 2; ++s)
                acc[mi][s] = __builtin_amdgcn_wmma_f32_16x16x32_bf16(
                    false, af[mi], false, bfr[s], (short)0, acc[mi][s], false, false);
        __syncthreads();
    }

    // epilogue; C/D layout: VGPR r -> M = +half*8 + r, N = +l15
    #pragma unroll
    for (int mi = 0; mi < 2; ++mi) {
        #pragma unroll
        for (int r = 0; r < 8; ++r) {
            const int gm = m0 + waveM * 32 + mi * 16 + half * 8 + r;
            #pragma unroll
            for (int s = 0; s < 2; ++s) {
                const int gn = n0 + waveN * 32 + s * 16 + l15;
                if (gn >= N) continue;
                float v = acc[mi][s][r];
                if (colBias) v += colBias[gn];
                if (rowBias) v += rowBias[gm];
                const size_t idx = (size_t)gm * (size_t)N + (size_t)gn;
                if (mode == 1)      v = Res[idx] + fmaxf(v, 0.f);
                else if (mode == 2) v = Res[idx] + v;
                C[idx] = v;
            }
        }
    }
}

// ---------------------------------------------------------------------------
// lhs_proj = lhs_embedding @ W_proj.T + b_proj   (256x128, tiny)
// ---------------------------------------------------------------------------
__global__ __launch_bounds__(128) void k_lhs_proj(
    const float* __restrict__ lhs, const float* __restrict__ Wp,
    const float* __restrict__ bp, float* __restrict__ out)
{
    const int m = blockIdx.x, n = threadIdx.x;
    float s = bp[n];
    #pragma unroll 4
    for (int k = 0; k < kC; ++k) s += lhs[m * kC + k] * Wp[n * kC + k];
    out[m * kC + n] = s;
}

// off_emb[m], off_id[m] = lhs_proj[m] . w + b
__global__ __launch_bounds__(128) void k_offsets(
    const float* __restrict__ lhsp,
    const float* __restrict__ we, const float* __restrict__ be,
    const float* __restrict__ wi, const float* __restrict__ bi,
    float* __restrict__ oe, float* __restrict__ oi)
{
    __shared__ float r1[128], r2[128];
    const int m = blockIdx.x, t = threadIdx.x;
    const float a = lhsp[m * kC + t];
    r1[t] = a * we[t];
    r2[t] = a * wi[t];
    __syncthreads();
    for (int s = 64; s > 0; s >>= 1) {
        if (t < s) { r1[t] += r1[t + s]; r2[t] += r2[t + s]; }
        __syncthreads();
    }
    if (t == 0) { oe[m] = r1[0] + be[0]; oi[m] = r2[0] + bi[0]; }
}

// ---------------------------------------------------------------------------
// inverse index map + idgnn scatter
// ---------------------------------------------------------------------------
__global__ void k_fill_inv(int* inv) {
    const int i = blockIdx.x * 256 + threadIdx.x;
    if (i < kNN) inv[i] = -1;
}
__global__ void k_scatter_inv(const int* __restrict__ ridx, int* __restrict__ inv) {
    const int i = blockIdx.x * 256 + threadIdx.x;
    if (i < kNID) inv[ridx[i]] = i;
}
__global__ void k_zero_counts(unsigned* c) { c[threadIdx.x] = 0u; }

// one wave32 per id-gnn row: two 128-length dots + scatter into logits
__global__ __launch_bounds__(256) void k_idgnn(
    const float* __restrict__ rhs_gnn, const float* __restrict__ lhs_emb,
    const int* __restrict__ ridx, const int* __restrict__ batch,
    const float* __restrict__ w_head, const float* __restrict__ b_head,
    const float* __restrict__ off_id, float* __restrict__ logits)
{
    const int wid  = threadIdx.x >> 5;
    const int lane = threadIdx.x & 31;
    const int i    = blockIdx.x * 8 + wid;
    if (i >= kNID) return;
    const int b = batch[i];
    const float* g = rhs_gnn + (size_t)i * kC;
    const float* l = lhs_emb + (size_t)b * kC;
    float d1 = 0.f, d2 = 0.f;
    for (int c = lane; c < kC; c += 32) {
        const float gv = g[c];
        d1 += gv * w_head[c];
        d2 += gv * l[c];
    }
    #pragma unroll
    for (int m = 16; m > 0; m >>= 1) {
        d1 += __shfl_xor(d1, m, 32);
        d2 += __shfl_xor(d2, m, 32);
    }
    if (lane == 0)
        logits[(size_t)b * kNN + ridx[i]] = d1 + b_head[0] + d2 + off_id[b];
}

// ---------------------------------------------------------------------------
// top-k: histogram threshold -> candidate collect -> bitonic sort
// ---------------------------------------------------------------------------
__global__ __launch_bounds__(256) void k_topk_hist(
    const float* __restrict__ logits, unsigned* __restrict__ thresh)
{
    __shared__ unsigned hist[4096];
    const int row = blockIdx.x, t = threadIdx.x;
    for (int i = t; i < 4096; i += 256) hist[i] = 0u;
    __syncthreads();
    const float* p = logits + (size_t)row * kNN;
    for (int j = t; j < kNN; j += 256)
        atomicAdd(&hist[mono(p[j]) >> 20], 1u);
    __syncthreads();
    if (t == 0) {
        unsigned cum = 0; int b = 0;
        for (b = 4095; b > 0; --b) { cum += hist[b]; if (cum >= kTOPK) break; }
        thresh[row] = (unsigned)b;
    }
}

__global__ __launch_bounds__(256) void k_topk_collect(
    const float* __restrict__ logits, const unsigned* __restrict__ thresh,
    unsigned* __restrict__ cnt, uint2* __restrict__ cand)
{
    const int row = blockIdx.x, t = threadIdx.x;
    const unsigned tb = thresh[row];
    const float* p = logits + (size_t)row * kNN;
    for (int j = t; j < kNN; j += 256) {
        const unsigned key = mono(p[j]);
        if ((key >> 20) >= tb) {
            const unsigned pos = atomicAdd(&cnt[row], 1u);
            if (pos < kCAP) cand[(size_t)row * kCAP + pos] = make_uint2(key, (unsigned)j);
        }
    }
}

__global__ __launch_bounds__(256) void k_topk_sort(
    const unsigned* __restrict__ cnt, const uint2* __restrict__ cand,
    int* __restrict__ topk)
{
    __shared__ unsigned skey[kCAP];
    __shared__ unsigned sidx[kCAP];
    const int row = blockIdx.x, t = threadIdx.x;
    unsigned c = cnt[row];
    if (c > kCAP) c = kCAP;
    for (int i = t; i < kCAP; i += 256) {
        if (i < (int)c) { const uint2 e = cand[(size_t)row * kCAP + i]; skey[i] = e.x; sidx[i] = e.y; }
        else            { skey[i] = 0u; sidx[i] = 0u; }
    }
    __syncthreads();
    for (int ksz = 2; ksz <= kCAP; ksz <<= 1) {
        for (int jj = ksz >> 1; jj > 0; jj >>= 1) {
            for (int i = t; i < kCAP; i += 256) {
                const int ixj = i ^ jj;
                if (ixj > i) {
                    const bool desc = ((i & ksz) == 0);
                    const unsigned ka = skey[i], kb2 = skey[ixj];
                    const bool sw = desc ? (ka < kb2) : (ka > kb2);
                    if (sw) {
                        skey[i] = kb2; skey[ixj] = ka;
                        const unsigned tmp = sidx[i]; sidx[i] = sidx[ixj]; sidx[ixj] = tmp;
                    }
                }
            }
            __syncthreads();
        }
    }
    if (t < kTOPK) topk[row * kTOPK + t] = (int)sidx[t];
}

// ---------------------------------------------------------------------------
// seq[b,k,:] = concat(gathered 128, lhs_uniq[b] 128)
// ---------------------------------------------------------------------------
__global__ __launch_bounds__(256) void k_build_seq(
    const int* __restrict__ topk, const int* __restrict__ inv,
    const float* __restrict__ rhs_gnn, const float* __restrict__ rhs_w,
    const float* __restrict__ lhsp, const int* __restrict__ batch,
    float* __restrict__ seq)
{
    const int row = blockIdx.x;                 // b*TOPK + k
    const int b   = row / kTOPK;
    const int t   = threadIdx.x;
    const int flat = topk[row];
    float v;
    if (t < kC) {
        const int pos = inv[flat];
        v = (pos >= 0) ? rhs_gnn[(size_t)pos * kC + t]
                       : rhs_w[(size_t)flat * kC + t];
    } else {
        v = lhsp[(size_t)batch[b] * kC + (t - kC)];
    }
    seq[(size_t)row * kD + t] = v;
}

// ---------------------------------------------------------------------------
// attention per (b, h): S=100, hd=64, scores in LDS, softmax, att@v
// ---------------------------------------------------------------------------
__global__ __launch_bounds__(256) void k_attn(
    const float* __restrict__ qkv, float* __restrict__ o)
{
    __shared__ float sc[kTOPK * kTOPK];
    const int b = blockIdx.x, h = blockIdx.y, t = threadIdx.x;
    const float scale = 0.125f;  // 1/sqrt(64)
    const float* base = qkv + (size_t)b * kTOPK * (3 * kD);

    for (int e = t; e < kTOPK * kTOPK; e += 256) {
        const int i = e / kTOPK, j = e % kTOPK;
        const float* q = base + i * (3 * kD) + h * kHD;
        const float* k = base + j * (3 * kD) + kD + h * kHD;
        float s = 0.f;
        #pragma unroll 8
        for (int d = 0; d < kHD; ++d) s += q[d] * k[d];
        sc[e] = s * scale;
    }
    __syncthreads();
    if (t < kTOPK) {
        float mx = -3.4e38f;
        for (int j = 0; j < kTOPK; ++j) mx = fmaxf(mx, sc[t * kTOPK + j]);
        float sum = 0.f;
        for (int j = 0; j < kTOPK; ++j) {
            const float e = __expf(sc[t * kTOPK + j] - mx);
            sc[t * kTOPK + j] = e; sum += e;
        }
        const float is = 1.f / sum;
        for (int j = 0; j < kTOPK; ++j) sc[t * kTOPK + j] *= is;
    }
    __syncthreads();
    for (int e = t; e < kTOPK * kHD; e += 256) {
        const int i = e >> 6, d = e & 63;
        const float* v = base + 2 * kD + h * kHD + d;
        float s = 0.f;
        for (int j = 0; j < kTOPK; ++j) s += sc[i * kTOPK + j] * v[(size_t)j * (3 * kD)];
        o[((size_t)b * kTOPK + i) * kD + h * kHD + d] = s;
    }
}

// ---------------------------------------------------------------------------
// LayerNorm over D=256 (in-place) and final tr_logits dot
// ---------------------------------------------------------------------------
__global__ __launch_bounds__(256) void k_ln(
    float* __restrict__ x, const float* __restrict__ g, const float* __restrict__ bb)
{
    __shared__ float r1[256], r2[256];
    const int row = blockIdx.x, t = threadIdx.x;
    const float v = x[(size_t)row * kD + t];
    r1[t] = v; r2[t] = v * v;
    __syncthreads();
    for (int s = 128; s > 0; s >>= 1) {
        if (t < s) { r1[t] += r1[t + s]; r2[t] += r2[t + s]; }
        __syncthreads();
    }
    const float mean = r1[0] * (1.f / kD);
    const float var  = r2[0] * (1.f / kD) - mean * mean;
    x[(size_t)row * kD + t] = (v - mean) * rsqrtf(var + 1e-5f) * g[t] + bb[t];
}

__global__ __launch_bounds__(256) void k_tr(
    const float* __restrict__ x, const float* __restrict__ w,
    const float* __restrict__ bb, float* __restrict__ out)
{
    __shared__ float r[256];
    const int row = blockIdx.x, t = threadIdx.x;
    r[t] = x[(size_t)row * kD + t] * w[t];
    __syncthreads();
    for (int s = 128; s > 0; s >>= 1) {
        if (t < s) r[t] += r[t + s];
        __syncthreads();
    }
    if (t == 0) out[row] = r[0] + bb[0];
}

// ---------------------------------------------------------------------------
// host launcher
// ---------------------------------------------------------------------------
extern "C" void kernel_launch(void* const* d_in, const int* in_sizes, int n_in,
                              void* d_out, int out_size, void* d_ws, size_t ws_size,
                              hipStream_t stream)
{
    (void)in_sizes; (void)n_in; (void)out_size; (void)ws_size;

    const float* lhs_emb   = (const float*)d_in[0];
    const float* rhs_gnn   = (const float*)d_in[1];
    const int*   ridx      = (const int*)  d_in[2];
    const int*   batch     = (const int*)  d_in[3];
    const float* rhs_w     = (const float*)d_in[4];
    const float* W_proj    = (const float*)d_in[5];
    const float* b_proj    = (const float*)d_in[6];
    const float* w_off_emb = (const float*)d_in[7];
    const float* b_off_emb = (const float*)d_in[8];
    const float* w_head    = (const float*)d_in[9];
    const float* b_head    = (const float*)d_in[10];
    const float* w_off_id  = (const float*)d_in[11];
    const float* b_off_id  = (const float*)d_in[12];
    const float* in_w      = (const float*)d_in[13];
    const float* in_b      = (const float*)d_in[14];
    const float* out_w     = (const float*)d_in[15];
    const float* out_b     = (const float*)d_in[16];
    const float* ln1_g     = (const float*)d_in[17];
    const float* ln1_b     = (const float*)d_in[18];
    const float* lin_w     = (const float*)d_in[19];
    const float* lin_b     = (const float*)d_in[20];
    const float* ln2_g     = (const float*)d_in[21];
    const float* ln2_b     = (const float*)d_in[22];
    const float* tr_w      = (const float*)d_in[23];
    const float* tr_b      = (const float*)d_in[24];

    // outputs: embgnn_logits (B*NN f32) | tr_logits (B*TOPK f32) | topk_idx (B*TOPK i32)
    float* emb_logits = (float*)d_out;
    float* tr_out     = emb_logits + (size_t)kB * kNN;
    int*   topk_out   = (int*)(tr_out + (size_t)kB * kTOPK);

    // workspace carve (aligned to 256B)
    char* w = (char*)d_ws;
    size_t off = 0;
    auto alloc = [&](size_t bytes) -> void* {
        void* p = w + off;
        off = (off + bytes + 255) & ~(size_t)255;
        return p;
    };
    float*    lhsp   = (float*)   alloc((size_t)kB * kC * 4);
    float*    off_e  = (float*)   alloc((size_t)kB * 4);
    float*    off_i  = (float*)   alloc((size_t)kB * 4);
    int*      inv    = (int*)     alloc((size_t)kNN * 4);
    unsigned* thresh = (unsigned*)alloc((size_t)kB * 4);
    unsigned* ccnt   = (unsigned*)alloc((size_t)kB * 4);
    uint2*    cand   = (uint2*)   alloc((size_t)kB * kCAP * 8);
    float*    seq    = (float*)   alloc((size_t)kROWS * kD * 4);
    float*    qkv    = (float*)   alloc((size_t)kROWS * 3 * kD * 4);
    float*    attn   = (float*)   alloc((size_t)kROWS * kD * 4);   // reused as x2
    float*    x1     = (float*)   alloc((size_t)kROWS * kD * 4);

    // 1) lhs_proj and offset scalars
    k_lhs_proj<<<kB, 128, 0, stream>>>(lhs_emb, W_proj, b_proj, lhsp);
    k_offsets<<<kB, 128, 0, stream>>>(lhsp, w_off_emb, b_off_emb, w_off_id, b_off_id, off_e, off_i);

    // 2) big WMMA GEMM: embgnn_logits = lhs_proj @ rhs_emb_weight.T + off_emb[:,None]
    k_gemm_bf16<<<dim3((kNN + 63) / 64, kB / 128), 256, 0, stream>>>(
        lhsp, rhs_w, off_e, nullptr, nullptr, emb_logits, kB, kNN, kC, 0);

    // 3) inverse map + idgnn scatter
    k_fill_inv<<<(kNN + 255) / 256, 256, 0, stream>>>(inv);
    k_scatter_inv<<<(kNID + 255) / 256, 256, 0, stream>>>(ridx, inv);
    k_zero_counts<<<1, 256, 0, stream>>>(ccnt);
    k_idgnn<<<(kNID + 7) / 8, 256, 0, stream>>>(
        rhs_gnn, lhs_emb, ridx, batch, w_head, b_head, off_i, emb_logits);

    // 4) top-k per row
    k_topk_hist<<<kB, 256, 0, stream>>>(emb_logits, thresh);
    k_topk_collect<<<kB, 256, 0, stream>>>(emb_logits, thresh, ccnt, cand);
    k_topk_sort<<<kB, 256, 0, stream>>>(ccnt, cand, topk_out);

    // 5) build seq and run transformer block (all GEMMs via WMMA)
    k_build_seq<<<kROWS, 256, 0, stream>>>(topk_out, inv, rhs_gnn, rhs_w, lhsp, batch, seq);

    k_gemm_bf16<<<dim3((3 * kD) / 64, kROWS / 128), 256, 0, stream>>>(
        seq, in_w, nullptr, in_b, nullptr, qkv, kROWS, 3 * kD, kD, 0);

    k_attn<<<dim3(kB, kHEADS), 256, 0, stream>>>(qkv, attn);

    // out = attn @ out_proj.T + out_b + seq (residual)
    k_gemm_bf16<<<dim3(kD / 64, kROWS / 128), 256, 0, stream>>>(
        attn, out_w, nullptr, out_b, seq, x1, kROWS, kD, kD, 2);

    k_ln<<<kROWS, 256, 0, stream>>>(x1, ln1_g, ln1_b);

    // out = x1 + relu(x1 @ lin_w.T + lin_b)  (write to attn buffer, now free)
    k_gemm_bf16<<<dim3(kD / 64, kROWS / 128), 256, 0, stream>>>(
        x1, lin_w, nullptr, lin_b, x1, attn, kROWS, kD, kD, 1);

    k_ln<<<kROWS, 256, 0, stream>>>(attn, ln2_g, ln2_b);

    k_tr<<<kROWS, 256, 0, stream>>>(attn, tr_w, tr_b, tr_out);
}
